// HRMChess_75977971466794
// MI455X (gfx1250) — matrix operand order repaired and checked by reading, not compiled, not run on hardware
//
#include <hip/hip_runtime.h>
#include <hip/hip_bf16.h>
#include <math.h>

// ---------------- types ----------------
typedef __attribute__((ext_vector_type(16))) _Float16 v16h;
typedef __attribute__((ext_vector_type(8)))  _Float16 v8h;
typedef __attribute__((ext_vector_type(8)))  float    v8f;
typedef int v4i __attribute__((vector_size(16)));   // matches async-LDS builtin param

#define DMODEL 256
#define SEQ    78
#define NHEAD  4
#define HD     64
#define BSZ    1024
#define MROWS  (BSZ * SEQ)      // 79872 = 128*624
#define MCHUNK 9984             // 78*128 ; MROWS = 8*MCHUNK
#define EPSLN  1e-5f
#define KC     128              // K-chunk staged into LDS (all K here are multiples of 128)

#define AS1 __attribute__((address_space(1)))
#define AS3 __attribute__((address_space(3)))

#if defined(__AMDGCN__) && \
    __has_builtin(__builtin_amdgcn_global_load_async_to_lds_b128) && \
    __has_builtin(__builtin_amdgcn_s_wait_asynccnt)
#define HAVE_ASYNC_LDS 1
#else
#define HAVE_ASYNC_LDS 0
#endif

// ---------------- weight convert + transpose: Wt[n*K + k] = (f16) W[k*N + n] ----------------
__global__ void k_transpose_cvt(const float* __restrict__ W, _Float16* __restrict__ Wt,
                                int K, int N) {
  int idx = blockIdx.x * 256 + threadIdx.x;
  if (idx >= K * N) return;
  int k = idx / N, n = idx % N;
  Wt[(size_t)n * K + k] = (_Float16)W[idx];
}

// ---------------- embedding gather: x[b,s,:] = emb[tok] + pos[s] ----------------
__global__ void k_embed(const int* __restrict__ fen, const int* __restrict__ uci,
                        const float* __restrict__ fw, const float* __restrict__ uw,
                        const float* __restrict__ pw, _Float16* __restrict__ x) {
  int row = blockIdx.x;            // 0 .. MROWS-1
  int d   = threadIdx.x;           // 0 .. 255
  int b = row / SEQ, s = row % SEQ;
  const float* tp = (s < 77) ? (fw + (size_t)fen[b * 77 + s] * DMODEL)
                             : (uw + (size_t)uci[b] * DMODEL);
  x[(size_t)row * DMODEL + d] = (_Float16)(tp[d] + pw[s * DMODEL + d]);
}

// ---------------- WMMA GEMM with async-LDS-staged B tile ----------------
// C[M,N] = act(A[M,K] @ W[K,N] + bias), W given transposed Wt[N,K] (row-major).
// block = 256 threads = 8 waves; wave -> 16(M) x 64(N) strip; block tile 128 x 64.
// B strip (64 cols x KC) is staged once per block into LDS (16 KB) via
// GLOBAL_LOAD_ASYNC_TO_LDS_B128 (ASYNCcnt) -- removes the 8x redundant per-wave
// global B streams. A fragments stay as direct global_load_b128 (no intra-block reuse).
// Requires M%128==0, N%64==0, K%KC==0.
template <bool RELU, typename OutT>
__global__ void k_wmma_gemm(const _Float16* __restrict__ A, const _Float16* __restrict__ Wt,
                            const float* __restrict__ bias, OutT* __restrict__ C,
                            int M, int N, int K) {
  __shared__ alignas(32) _Float16 Bs[64 * KC];   // 16 KB

  const int tid   = threadIdx.x;
  const int lane  = tid & 31;
  const int wave  = tid >> 5;
  const int row16 = lane & 15;
  const int half  = lane >> 4;          // 0: lanes 0-15, 1: lanes 16-31
  const int kb    = half * 16;          // A frag K sub-base
  const int koff  = half * 16;          // B frag K sub-base

  const int mBase = blockIdx.y * 128 + wave * 16;
  const int nBase = blockIdx.x * 64;

  v8f acc0 = {}, acc1 = {}, acc2 = {}, acc3 = {};
  const _Float16* arow = A + (size_t)(mBase + row16) * K;

  for (int kBase = 0; kBase < K; kBase += KC) {
    __syncthreads();  // previous chunk fully consumed before overwrite
    // ---- stage B chunk: 64 rows (output cols) x KC f16 = 16 KB; 4 x b128 per thread ----
#if HAVE_ASYNC_LDS
#pragma unroll
    for (int it = 0; it < 4; ++it) {
      int idx = it * 256 + tid;               // 0..1023 chunks of 8 f16 (16 B)
      int r = idx >> 4, ck = (idx & 15) * 8;  // 16 chunks per row of KC=128
      const _Float16* gp = Wt + (size_t)(nBase + r) * K + kBase + ck;
      _Float16* lp = &Bs[r * KC + ck];
      __builtin_amdgcn_global_load_async_to_lds_b128((AS1 v4i*)gp, (AS3 v4i*)lp, 0, 0);
    }
    __builtin_amdgcn_s_wait_asynccnt(0);
#else
#pragma unroll
    for (int it = 0; it < 4; ++it) {
      int idx = it * 256 + tid;
      int r = idx >> 4, ck = (idx & 15) * 8;
      *(v8h*)&Bs[r * KC + ck] = *(const v8h*)(Wt + (size_t)(nBase + r) * K + kBase + ck);
    }
#endif
    __syncthreads();

    // ---- 4 K-steps of 32 against the staged chunk ----
#pragma unroll
    for (int kk = 0; kk < KC; kk += 32) {
      const int k0 = kBase + kk;
      // A fragment (16x32 f16): lane holds A[row, k0+kb+0..15]
      v8h alo = *(const v8h*)(arow + k0 + kb);
      v8h ahi = *(const v8h*)(arow + k0 + kb + 8);
      v16h af;
#pragma unroll
      for (int j = 0; j < 8; ++j) { af[j] = alo[j]; af[8 + j] = ahi[j]; }
      // B fragments from LDS: lane holds Bs[col][kk+koff+0..15] (32 B contiguous)
      const _Float16* bb = &Bs[kk + koff];
      v16h bf0 = *(const v16h*)(bb + (row16 +  0) * KC);
      v16h bf1 = *(const v16h*)(bb + (row16 + 16) * KC);
      v16h bf2 = *(const v16h*)(bb + (row16 + 32) * KC);
      v16h bf3 = *(const v16h*)(bb + (row16 + 48) * KC);
      acc0 = __builtin_amdgcn_wmma_f32_16x16x32_f16(false, af, false, bf0, (short)0, acc0, false, false);
      acc1 = __builtin_amdgcn_wmma_f32_16x16x32_f16(false, af, false, bf1, (short)0, acc1, false, false);
      acc2 = __builtin_amdgcn_wmma_f32_16x16x32_f16(false, af, false, bf2, (short)0, acc2, false, false);
      acc3 = __builtin_amdgcn_wmma_f32_16x16x32_f16(false, af, false, bf3, (short)0, acc3, false, false);
    }
  }

  // C/D layout: lane L, elem v -> M = mBase + v + 8*half, N = nBase + t*16 + (L%16)
  v8f accs[4] = {acc0, acc1, acc2, acc3};
#pragma unroll
  for (int t = 0; t < 4; ++t) {
    int col = nBase + t * 16 + row16;
    float bv = bias ? bias[col] : 0.f;
#pragma unroll
    for (int v = 0; v < 8; ++v) {
      float r = accs[t][v] + bv;
      if (RELU) r = fmaxf(r, 0.f);
      C[(size_t)(mBase + v + 8 * half) * N + col] = (OutT)r;
    }
  }
}

// ---------------- fused attention: scores + softmax + AV, one wave per (b,h,q) ----------------
__global__ void k_attention(const _Float16* __restrict__ qkv, _Float16* __restrict__ o) {
  __shared__ float lq[8][HD];
  __shared__ float pb[8][80];
  const int wv = threadIdx.x >> 5, lane = threadIdx.x & 31;
  const int gid = blockIdx.x * 8 + wv;
  const int b = gid / (NHEAD * SEQ);
  const int r = gid % (NHEAD * SEQ);
  const int h = r / SEQ, q = r % SEQ;
  const _Float16* base = qkv + (size_t)b * SEQ * (3 * DMODEL);

  // stage q row into LDS (f32)
  const _Float16* qp = base + (size_t)q * (3 * DMODEL) + h * HD;
  for (int j = lane; j < HD; j += 32) lq[wv][j] = (float)qp[j];
  __syncthreads();

  // scores for keys k = lane, lane+32, lane+64
  float sc[3];
  float mx = -1e30f;
#pragma unroll
  for (int t = 0; t < 3; ++t) {
    int k = lane + t * 32;
    float s = -1e30f;
    if (k < SEQ) {
      const _Float16* kp = base + (size_t)k * (3 * DMODEL) + DMODEL + h * HD;
      s = 0.f;
#pragma unroll 8
      for (int d = 0; d < HD; ++d) s += lq[wv][d] * (float)kp[d];
      s *= 0.125f;  // 1/sqrt(64)
    }
    sc[t] = s;
    mx = fmaxf(mx, s);
  }
#pragma unroll
  for (int off = 16; off; off >>= 1) mx = fmaxf(mx, __shfl_xor(mx, off, 32));
  float e[3], se = 0.f;
#pragma unroll
  for (int t = 0; t < 3; ++t) {
    e[t] = (sc[t] > -1e29f) ? __expf(sc[t] - mx) : 0.f;
    se += e[t];
  }
#pragma unroll
  for (int off = 16; off; off >>= 1) se += __shfl_xor(se, off, 32);
  float inv = 1.f / se;
#pragma unroll
  for (int t = 0; t < 3; ++t) {
    int k = lane + t * 32;
    if (k < SEQ) pb[wv][k] = e[t] * inv;
  }
  __syncthreads();

  // out[d] = sum_k p[k] * V[k,d]; lane covers dims 2*lane, 2*lane+1
  const int d0 = lane * 2;
  float a0 = 0.f, a1 = 0.f;
  const _Float16* vp = base + 2 * DMODEL + h * HD;
  for (int k = 0; k < SEQ; ++k) {
    float pk = pb[wv][k];
    a0 += pk * (float)vp[(size_t)k * (3 * DMODEL) + d0];
    a1 += pk * (float)vp[(size_t)k * (3 * DMODEL) + d0 + 1];
  }
  _Float16* op = o + (size_t)(b * SEQ + q) * DMODEL + h * HD;
  op[d0] = (_Float16)a0;
  op[d0 + 1] = (_Float16)a1;
}

// ---------------- layernorm over D=256 (optional residual), one wave per row ----------------
__global__ void k_residual_ln(const _Float16* __restrict__ resid, const _Float16* __restrict__ y,
                              const float* __restrict__ g, const float* __restrict__ b,
                              _Float16* __restrict__ out, int rows) {
  const int wv = threadIdx.x >> 5, lane = threadIdx.x & 31;
  const int row = blockIdx.x * 8 + wv;
  if (row >= rows) return;
  const size_t off = (size_t)row * DMODEL + lane * 8;
  v8h yv = *(const v8h*)(y + off);
  float v[8];
  float s = 0.f, sq = 0.f;
  if (resid) {
    v8h rv = *(const v8h*)(resid + off);
#pragma unroll
    for (int j = 0; j < 8; ++j) v[j] = (float)yv[j] + (float)rv[j];
  } else {
#pragma unroll
    for (int j = 0; j < 8; ++j) v[j] = (float)yv[j];
  }
#pragma unroll
  for (int j = 0; j < 8; ++j) { s += v[j]; sq += v[j] * v[j]; }
#pragma unroll
  for (int o2 = 16; o2; o2 >>= 1) { s += __shfl_xor(s, o2, 32); sq += __shfl_xor(sq, o2, 32); }
  const float mean = s * (1.f / DMODEL);
  const float var  = sq * (1.f / DMODEL) - mean * mean;
  const float rstd = rsqrtf(var + EPSLN);
  v8h ov;
#pragma unroll
  for (int j = 0; j < 8; ++j) {
    int d = lane * 8 + j;
    ov[j] = (_Float16)((v[j] - mean) * rstd * g[d] + b[d]);
  }
  *(v8h*)(out + off) = ov;
}

// ---------------- mean over seq: [B,78,256] -> [B,256] ----------------
__global__ void k_meanpool(const _Float16* __restrict__ in, _Float16* __restrict__ out) {
  int idx = blockIdx.x * 256 + threadIdx.x;   // B*256 threads
  int b = idx >> 8, d = idx & 255;
  float s = 0.f;
  for (int t = 0; t < SEQ; ++t) s += (float)in[(size_t)(b * SEQ + t) * DMODEL + d];
  out[idx] = (_Float16)(s * (1.f / SEQ));
}

// ---------------- strided 2D f16 copy (concat slices) ----------------
__global__ void k_copy2d(_Float16* __restrict__ dst, int dpitch,
                         const _Float16* __restrict__ src, int spitch,
                         int rows, int cols) {
  int idx = blockIdx.x * 256 + threadIdx.x;
  if (idx >= rows * cols) return;
  int r = idx / cols, c = idx % cols;
  dst[(size_t)r * dpitch + c] = src[(size_t)r * spitch + c];
}

__global__ void k_zero16(_Float16* __restrict__ p, int n) {
  int idx = blockIdx.x * 256 + threadIdx.x;
  if (idx < n) p[idx] = (_Float16)0.f;
}

// ================= host orchestration =================
static void gemm16(hipStream_t st, const _Float16* A, const _Float16* Wt, const float* bias,
                   _Float16* C, int M, int N, int K, bool relu) {
  dim3 g(N / 64, M / 128), blk(256);
  if (relu) k_wmma_gemm<true, _Float16><<<g, blk, 0, st>>>(A, Wt, bias, C, M, N, K);
  else      k_wmma_gemm<false, _Float16><<<g, blk, 0, st>>>(A, Wt, bias, C, M, N, K);
}

extern "C" void kernel_launch(void* const* d_in, const int* in_sizes, int n_in,
                              void* d_out, int out_size, void* d_ws, size_t ws_size,
                              hipStream_t stream) {
  (void)in_sizes; (void)n_in; (void)out_size; (void)ws_size;

  // ---- inputs (setup_inputs order) ----
  const int*   fen   = (const int*)d_in[0];
  const int*   uci   = (const int*)d_in[1];
  const float* fenw  = (const float*)d_in[2];
  const float* uciw  = (const float*)d_in[3];
  const float* posw  = (const float*)d_in[4];
  const float* wqkv  = (const float*)d_in[5];
  const float* bqkv  = (const float*)d_in[6];
  const float* wo    = (const float*)d_in[7];
  const float* bo    = (const float*)d_in[8];
  const float* ln1g  = (const float*)d_in[9];
  const float* ln1b  = (const float*)d_in[10];
  const float* w1    = (const float*)d_in[11];
  const float* b1    = (const float*)d_in[12];
  const float* w2    = (const float*)d_in[13];
  const float* b2    = (const float*)d_in[14];
  const float* ln2g  = (const float*)d_in[15];
  const float* ln2b  = (const float*)d_in[16];
  const float* postg = (const float*)d_in[17];
  const float* postb = (const float*)d_in[18];
  const float* pw1   = (const float*)d_in[19];
  const float* pb1   = (const float*)d_in[20];
  const float* pw2   = (const float*)d_in[21];
  const float* pb2   = (const float*)d_in[22];
  const float* Lw1   = (const float*)d_in[23];
  const float* Lb1   = (const float*)d_in[24];
  const float* Lw2   = (const float*)d_in[25];
  const float* Lb2   = (const float*)d_in[26];
  const float* Llng  = (const float*)d_in[27];
  const float* Llnb  = (const float*)d_in[28];
  const float* Hw1   = (const float*)d_in[29];
  const float* Hb1   = (const float*)d_in[30];
  const float* Hw2   = (const float*)d_in[31];
  const float* Hb2   = (const float*)d_in[32];
  const float* Hlng  = (const float*)d_in[33];
  const float* Hlnb  = (const float*)d_in[34];
  const float* vhg   = (const float*)d_in[35];
  const float* vhb   = (const float*)d_in[36];
  const float* vw1   = (const float*)d_in[37];
  const float* vb1   = (const float*)d_in[38];
  const float* vw2   = (const float*)d_in[39];
  const float* vb2   = (const float*)d_in[40];

  // ---- workspace layout (bump allocator, 256B aligned) ----
  size_t off = 0;
  auto a16 = [&](size_t elems) -> _Float16* {
    _Float16* p = (_Float16*)((char*)d_ws + off);
    off += ((elems * sizeof(_Float16)) + 255) & ~(size_t)255;
    return p;
  };
  _Float16* wt_qkv = a16((size_t)4 * 256 * 768);
  _Float16* wt_wo  = a16((size_t)4 * 256 * 256);
  _Float16* wt_w1  = a16((size_t)4 * 256 * 2048);
  _Float16* wt_w2  = a16((size_t)4 * 2048 * 256);
  _Float16* wt_p1  = a16(256 * 256);
  _Float16* wt_p2  = a16(256 * 256);
  _Float16* wt_L1  = a16(768 * 256);
  _Float16* wt_L2  = a16(256 * 256);
  _Float16* wt_H1  = a16(512 * 256);
  _Float16* wt_H2  = a16(256 * 256);
  _Float16* wt_v1  = a16(256 * 256);
  _Float16* wt_v2  = a16(256 * 128);
  _Float16* xbuf   = a16((size_t)MROWS * 256);
  _Float16* qkvb   = a16((size_t)MROWS * 768);
  _Float16* obuf   = a16((size_t)MROWS * 256);
  _Float16* ybuf   = a16((size_t)MROWS * 256);
  _Float16* hbuf   = a16((size_t)MCHUNK * 2048);
  _Float16* outv   = a16((size_t)BSZ * 256);
  _Float16* zL     = a16((size_t)BSZ * 256);
  _Float16* zH     = a16((size_t)BSZ * 256);
  _Float16* catL   = a16((size_t)BSZ * 768);
  _Float16* catH   = a16((size_t)BSZ * 512);
  _Float16* t1     = a16((size_t)BSZ * 256);
  _Float16* t2     = a16((size_t)BSZ * 256);

  auto tc = [&](const float* W, _Float16* Wt, int K, int N) {
    int n = K * N;
    k_transpose_cvt<<<(n + 255) / 256, 256, 0, stream>>>(W, Wt, K, N);
  };
  auto ln = [&](const _Float16* res, const _Float16* y, const float* g, const float* b,
                _Float16* out, int rows) {
    k_residual_ln<<<rows / 8, 256, 0, stream>>>(res, y, g, b, out, rows);
  };
  auto cp = [&](_Float16* dst, int dp, const _Float16* src, int sp, int rows, int cols) {
    int n = rows * cols;
    k_copy2d<<<(n + 255) / 256, 256, 0, stream>>>(dst, dp, src, sp, rows, cols);
  };

  // ---- stage weights: fp32 -> f16 transposed ----
  for (int l = 0; l < 4; ++l) {
    tc(wqkv + (size_t)l * 256 * 768, wt_qkv + (size_t)l * 256 * 768, 256, 768);
    tc(wo   + (size_t)l * 256 * 256, wt_wo  + (size_t)l * 256 * 256, 256, 256);
    tc(w1   + (size_t)l * 256 * 2048, wt_w1 + (size_t)l * 256 * 2048, 256, 2048);
    tc(w2   + (size_t)l * 2048 * 256, wt_w2 + (size_t)l * 2048 * 256, 2048, 256);
  }
  tc(pw1, wt_p1, 256, 256);  tc(pw2, wt_p2, 256, 256);
  tc(Lw1, wt_L1, 768, 256);  tc(Lw2, wt_L2, 256, 256);
  tc(Hw1, wt_H1, 512, 256);  tc(Hw2, wt_H2, 256, 256);
  tc(vw1, wt_v1, 256, 256);  tc(vw2, wt_v2, 256, 128);

  // ---- embed ----
  k_embed<<<MROWS, 256, 0, stream>>>(fen, uci, fenw, uciw, posw, xbuf);

  // ---- 4 transformer layers ----
  for (int l = 0; l < 4; ++l) {
    gemm16(stream, xbuf, wt_qkv + (size_t)l * 256 * 768, bqkv + l * 768, qkvb,
           MROWS, 768, 256, false);
    k_attention<<<(BSZ * NHEAD * SEQ) / 8, 256, 0, stream>>>(qkvb, obuf);
    gemm16(stream, obuf, wt_wo + (size_t)l * 256 * 256, bo + l * 256, ybuf,
           MROWS, 256, 256, false);
    ln(xbuf, ybuf, ln1g + l * 256, ln1b + l * 256, xbuf, MROWS);
    for (int c = 0; c < 8; ++c) {
      const _Float16* Ax = xbuf + (size_t)c * MCHUNK * 256;
      gemm16(stream, Ax, wt_w1 + (size_t)l * 256 * 2048, b1 + l * 2048, hbuf,
             MCHUNK, 2048, 256, true);
      gemm16(stream, hbuf, wt_w2 + (size_t)l * 2048 * 256, b2 + l * 256,
             ybuf + (size_t)c * MCHUNK * 256, MCHUNK, 256, 2048, false);
    }
    ln(xbuf, ybuf, ln2g + l * 256, ln2b + l * 256, xbuf, MROWS);
  }

  // ---- post-LN, pool MLP, mean over seq ----
  ln(nullptr, xbuf, postg, postb, xbuf, MROWS);
  gemm16(stream, xbuf, wt_p1, pb1, ybuf, MROWS, 256, 256, true);
  gemm16(stream, ybuf, wt_p2, pb2, obuf, MROWS, 256, 256, true);
  k_meanpool<<<(BSZ * 256) / 256, 256, 0, stream>>>(obuf, outv);

  // ---- HRM recurrence ----
  int nz = BSZ * 256;
  k_zero16<<<(nz + 255) / 256, 256, 0, stream>>>(zL, nz);
  k_zero16<<<(nz + 255) / 256, 256, 0, stream>>>(zH, nz);
  cp(catL + 512, 768, outv, 256, BSZ, 256);  // constant "out" slice

  auto l_net = [&]() {
    cp(catL + 0, 768, zL, 256, BSZ, 256);
    cp(catL + 256, 768, zH, 256, BSZ, 256);
    gemm16(stream, catL, wt_L1, Lb1, t1, BSZ, 256, 768, true);
    gemm16(stream, t1, wt_L2, Lb2, t2, BSZ, 256, 256, true);
    ln(nullptr, t2, Llng, Llnb, zL, BSZ);
  };
  auto h_net = [&]() {
    cp(catH + 0, 512, zH, 256, BSZ, 256);
    cp(catH + 256, 512, zL, 256, BSZ, 256);
    gemm16(stream, catH, wt_H1, Hb1, t1, BSZ, 256, 512, true);
    gemm16(stream, t1, wt_H2, Hb2, t2, BSZ, 256, 256, true);
    ln(nullptr, t2, Hlng, Hlnb, zH, BSZ);
  };

  for (int i = 0; i < 63; ++i) {
    l_net();
    if (((i + 1) % 8) == 0) h_net();  // zH only observable on keep-steps
  }
  l_net();  // final explicit step
  h_net();

  // ---- value head -> f32 output [1024, 128] ----
  ln(nullptr, zH, vhg, vhb, t2, BSZ);
  gemm16(stream, t2, wt_v1, vb1, t1, BSZ, 256, 256, true);
  {
    dim3 g(128 / 64, BSZ / 128), blk(256);
    k_wmma_gemm<false, float><<<g, blk, 0, stream>>>(t1, wt_v2, vb2, (float*)d_out,
                                                     BSZ, 128, 256);
  }
}
